// Toy_35089882808512
// MI455X (gfx1250) — matrix operand order repaired and checked by reading, not compiled, not run on hardware
//
#include <hip/hip_runtime.h>

// ---------------------------------------------------------------------------
// Problem constants (from reference): V=50257, D=512, T=256, K=256
// ---------------------------------------------------------------------------
constexpr int kV = 50257;
constexpr int kD = 512;
constexpr int kT = 256;
constexpr int kK = 256;

typedef __bf16 bf16_t;
typedef __bf16 v16bf __attribute__((ext_vector_type(16)));
typedef float  v8f   __attribute__((ext_vector_type(8)));

union BFrag {
  v16bf v;
  uint4 q[2];
};

// LDS byte offset of a __shared__ object (addrspacecast generic -> AS3)
__device__ __forceinline__ uint32_t lds_off(void* p) {
  return (uint32_t)(unsigned long long)(__attribute__((address_space(3))) void*)p;
}

// ---------------------------------------------------------------------------
// Kernel 0: zero the global accumulators (S1[256] + sumsq[1], contiguous)
// ---------------------------------------------------------------------------
__global__ void zero_acc(float* __restrict__ p, int n) {
  int i = blockIdx.x * blockDim.x + threadIdx.x;
  if (i < n) p[i] = 0.0f;
}

// ---------------------------------------------------------------------------
// Kernel 1: precompute (fp32 dot products, bf16 outputs)
//   Mt[t][d] = sum_k B[d][k] * t2v[t][k]        (B @ t2v^T, transposed)
//   Qt[d][t] = sum_k A[d][k] * t2v[t][k]        (t2v @ A^T, transposed)
// Layouts chosen so WMMA B-fragments are two contiguous 8-element k runs.
// ---------------------------------------------------------------------------
__global__ __launch_bounds__(256) void precompute(
    const float* __restrict__ B, const float* __restrict__ A,
    const float* __restrict__ t2v,
    bf16_t* __restrict__ Mt, bf16_t* __restrict__ Qt) {
  int idx = blockIdx.x * 256 + threadIdx.x;
  if (idx < kT * kD) {                       // Mt
    int t = idx >> 9;                        // / kD
    int d = idx & (kD - 1);
    const float* brow = B + (size_t)d * kK;
    const float* trow = t2v + (size_t)t * kK;
    float s = 0.0f;
    for (int k = 0; k < kK; ++k) s = fmaf(brow[k], trow[k], s);
    Mt[(size_t)t * kD + d] = (bf16_t)s;
  } else {                                   // Qt
    int i2 = idx - kT * kD;
    int d = i2 >> 8;                         // / kT
    int t = i2 & (kT - 1);
    const float* arow = A + (size_t)d * kK;
    const float* trow = t2v + (size_t)t * kK;
    float s = 0.0f;
    for (int k = 0; k < kK; ++k) s = fmaf(arow[k], trow[k], s);
    Qt[(size_t)d * kT + t] = (bf16_t)s;
  }
}

// ---------------------------------------------------------------------------
// Kernel 2: fused main pass over 16-row vocab tiles.
// 128 threads = 4 waves per workgroup.
// ---------------------------------------------------------------------------
__global__ __launch_bounds__(128) void fused_main(
    const float* __restrict__ w2v,
    const bf16_t* __restrict__ Mt,     // [256][512]
    const bf16_t* __restrict__ Qt,     // [512][256]
    const float* __restrict__ sigma,
    float* __restrict__ outAlpha,      // [T][V]
    float* __restrict__ outP,          // [V]
    float* __restrict__ outST,         // [T][V]
    float* __restrict__ gS1,           // [256] per-topic alpha sums
    float* __restrict__ gSumSq) {      // [1]   sum alpha^2
  // Padded strides break LDS bank conflicts on column-wise reads.
  __shared__ float  wTile[16][516];    // fp32 staged w2v tile (for ||w-mu||^2)
  __shared__ bf16_t wBf[16][520];      // bf16 copy: GEMM1 A operand
  __shared__ float  sTile[16][260];    // fp32 logits
  __shared__ bf16_t aTile[16][264];    // bf16 alpha: GEMM2 A operand
  __shared__ float  redA[16][8];
  __shared__ float  rowMax[16], rowSumInv[16];
  __shared__ float  topicSum[kT];
  __shared__ float  red1[128];
  __shared__ float  redP[128][8];

  const int tid  = threadIdx.x;
  const int lane = tid & 31;
  const int wave = tid >> 5;
  const int hlf  = lane >> 4;          // 0: lanes 0-15, 1: lanes 16-31
  const int l16  = lane & 15;
  const int v0   = blockIdx.x * 16;
  const float sig = sigma[0];

  for (int i = tid; i < kT; i += 128) topicSum[i] = 0.0f;

  // ---- stage w2v tile (16 x 512 fp32) via async global->LDS copies ----
  for (int i = tid; i < 2048; i += 128) {   // 2048 x 16B
    int r = i >> 7;                          // 128 float4 per row
    int c = i & 127;
    int v = v0 + r;
    if (v < kV) {
      uint32_t lo = lds_off(&wTile[r][0]) + (uint32_t)c * 16u;
      unsigned long long ga =
          (unsigned long long)(w2v + (size_t)v * kD + (size_t)c * 4);
      asm volatile("global_load_async_to_lds_b128 %0, %1, off"
                   :: "v"(lo), "v"(ga) : "memory");
    } else {
      ((float4*)&wTile[r][0])[c] = make_float4(0.f, 0.f, 0.f, 0.f);
    }
  }
  asm volatile("s_wait_asynccnt 0" ::: "memory");
  __syncthreads();

  // ---- one-shot fp32 -> bf16 conversion of the tile (shared by all waves)
  for (int i = tid; i < 1024; i += 128) {    // 8192 floats / 8
    int r = i >> 6;
    int c = (i & 63) * 8;
    const float* srcp = &wTile[r][c];
    bf16_t tmp[8];
#pragma unroll
    for (int j = 0; j < 8; ++j) tmp[j] = (bf16_t)srcp[j];
    *(uint4*)&wBf[r][c] = *(const uint4*)tmp;
  }
  __syncthreads();

  // ---- GEMM1: s[16][256] = w[16][512] @ M[512][256] (bf16 WMMA, fp32 acc)
  v8f acc[4];
  const v8f vzero = {0.f, 0.f, 0.f, 0.f, 0.f, 0.f, 0.f, 0.f};
#pragma unroll
  for (int q = 0; q < 4; ++q) acc[q] = vzero;

  for (int kb = 0; kb < kD; kb += 32) {
    // A fragment: row m = l16; K runs {kb+8h..+7} and {kb+16+8h..+7}
    BFrag a;
    {
      const bf16_t* row = &wBf[l16][0];
      const int k0 = kb + hlf * 8;
      a.q[0] = *(const uint4*)(row + k0);
      a.q[1] = *(const uint4*)(row + k0 + 16);
    }
#pragma unroll
    for (int q = 0; q < 4; ++q) {
      const int trow = wave * 64 + q * 16 + l16;           // N column (topic)
      const bf16_t* src = Mt + (size_t)trow * kD + kb + hlf * 8;
      __builtin_prefetch((const void*)(src + 32), 0, 1);   // next K block
      BFrag b;
      b.q[0] = *(const uint4*)(src);
      b.q[1] = *(const uint4*)(src + 16);
      acc[q] = __builtin_amdgcn_wmma_f32_16x16x32_bf16(
          false, a.v, false, b.v, (short)0, acc[q], false, false);
    }
  }

  // scatter s accumulators to LDS: (m = j + 8*hlf, n = topic column)
#pragma unroll
  for (int q = 0; q < 4; ++q) {
    const int n = wave * 64 + q * 16 + l16;
#pragma unroll
    for (int j = 0; j < 8; ++j) sTile[j + hlf * 8][n] = acc[q][j];
  }
  __syncthreads();

  // ---- softmax over T=256 per row (fp32), + RL partials ----
  const int r = tid & 15;
  const int c = tid >> 4;              // 8 chunks of 32 topics
  const int vRow = v0 + r;
  const bool valid = vRow < kV;

  float m = -3.4e38f;
  for (int j = 0; j < 32; ++j) m = fmaxf(m, sTile[r][c * 32 + j]);
  redA[r][c] = m;
  __syncthreads();
  if (tid < 16) {
    float mm = redA[tid][0];
    for (int j = 1; j < 8; ++j) mm = fmaxf(mm, redA[tid][j]);
    rowMax[tid] = mm;
  }
  __syncthreads();
  const float rm = rowMax[r];
  float ps = 0.0f;
  for (int j = 0; j < 32; ++j) ps += __expf(sTile[r][c * 32 + j] - rm);
  redA[r][c] = ps;
  __syncthreads();
  if (tid < 16) {
    float ss = 0.0f;
    for (int j = 0; j < 8; ++j) ss += redA[tid][j];
    rowSumInv[tid] = 1.0f / ss;
  }
  __syncthreads();
  const float inv = rowSumInv[r];
  float sq = 0.0f;
  for (int j = 0; j < 32; ++j) {
    const int t = c * 32 + j;
    const float sval = sTile[r][t];
    const float a = __expf(sval - rm) * inv;
    if (valid) {
      aTile[r][t] = (bf16_t)a;
      outAlpha[(size_t)t * kV + vRow] = a;    // alpha^T output
      outST[(size_t)t * kV + vRow]    = sval; // s^T output
      sq += a * a;
      atomicAdd(&topicSum[t], a);
    } else {
      aTile[r][t] = (bf16_t)0.0f;             // pad rows contribute nothing
    }
  }

  // block-reduce sum(alpha^2), publish RL partials
  red1[tid] = sq;
  __syncthreads();
  for (int s2 = 64; s2 > 0; s2 >>= 1) {
    if (tid < s2) red1[tid] += red1[tid + s2];
    __syncthreads();
  }
  if (tid == 0) atomicAdd(gSumSq, red1[0]);
  for (int t = tid; t < kT; t += 128) atomicAdd(&gS1[t], topicSum[t]);

  // ---- GEMM2: mu[16][512] = alpha[16][256] @ Q[256][512]; fold ||w-mu||^2 ----
  v8f macc[8];
  float pacc[8];
#pragma unroll
  for (int dt = 0; dt < 8; ++dt) { macc[dt] = vzero; pacc[dt] = 0.0f; }

  for (int kb = 0; kb < kT; kb += 32) {
    BFrag a;                                    // bf16 alpha, direct LDS loads
    const bf16_t* row = &aTile[l16][0];
    const int k0 = kb + hlf * 8;
    a.q[0] = *(const uint4*)(row + k0);
    a.q[1] = *(const uint4*)(row + k0 + 16);
#pragma unroll
    for (int dt = 0; dt < 8; ++dt) {
      const int dcol = wave * 128 + dt * 16 + l16;
      const bf16_t* src = Qt + (size_t)dcol * kT + kb + hlf * 8;
      BFrag b;
      b.q[0] = *(const uint4*)(src);
      b.q[1] = *(const uint4*)(src + 16);
      macc[dt] = __builtin_amdgcn_wmma_f32_16x16x32_bf16(
          false, a.v, false, b.v, (short)0, macc[dt], false, false);
    }
  }
#pragma unroll
  for (int dt = 0; dt < 8; ++dt) {
    const int dcol = wave * 128 + dt * 16 + l16;
#pragma unroll
    for (int j = 0; j < 8; ++j) {               // row m = j + 8*hlf
      const float diff = wTile[j + hlf * 8][dcol] - macc[dt][j];
      pacc[j] = fmaf(diff, diff, pacc[j]);
    }
  }
  // deterministic reduction of per-lane partials -> P
#pragma unroll
  for (int j = 0; j < 8; ++j) redP[tid][j] = pacc[j];
  __syncthreads();
  if (tid < 16) {                               // tid = row m
    const int mh = tid >> 3, mj = tid & 7;
    float s = 0.0f;
    for (int w = 0; w < 4; ++w)
      for (int l = 0; l < 16; ++l) s += redP[w * 32 + mh * 16 + l][mj];
    const int v = v0 + tid;
    if (v < kV) outP[v] = s / sig;
  }
}

// ---------------------------------------------------------------------------
// Kernel 3: RL = sum(alpha^2) - (1/V) * sum_t S1[t]^2 ; copy sigma
// ---------------------------------------------------------------------------
__global__ __launch_bounds__(256) void finalize(
    const float* __restrict__ gS1, const float* __restrict__ gSumSq,
    const float* __restrict__ sigma,
    float* __restrict__ outRL, float* __restrict__ outSigma) {
  __shared__ float red[256];
  const int t = threadIdx.x;
  const float s1 = gS1[t];
  red[t] = s1 * s1;
  __syncthreads();
  for (int s = 128; s > 0; s >>= 1) {
    if (t < s) red[t] += red[t + s];
    __syncthreads();
  }
  if (t == 0) {
    *outRL = *gSumSq - red[0] / (float)kV;
    *outSigma = sigma[0];
  }
}

// ---------------------------------------------------------------------------
// Host launcher
// ---------------------------------------------------------------------------
extern "C" void kernel_launch(void* const* d_in, const int* in_sizes, int n_in,
                              void* d_out, int out_size, void* d_ws, size_t ws_size,
                              hipStream_t stream) {
  (void)in_sizes; (void)n_in; (void)out_size; (void)ws_size;
  const float* w2v   = (const float*)d_in[0];   // [V][D]
  const float* t2v   = (const float*)d_in[1];   // [T][K]
  const float* B     = (const float*)d_in[2];   // [D][K]
  const float* A     = (const float*)d_in[3];   // [D][K]
  const float* sigma = (const float*)d_in[4];   // [1]

  // workspace layout
  char* ws = (char*)d_ws;
  bf16_t* Mt     = (bf16_t*)ws;                            // 256*512*2 = 256KB
  bf16_t* Qt     = (bf16_t*)(ws + 262144);                 // 512*256*2 = 256KB
  float*  gS1    = (float*)(ws + 524288);                  // 256 floats
  float*  gSumSq = (float*)(ws + 524288 + 1024);           // 1 float

  // output layout: alpha [T*V], P [V], RL [1], s^T [T*V], sigma [1]
  float* out      = (float*)d_out;
  float* outAlpha = out;
  float* outP     = out + (size_t)kT * kV;
  float* outRL    = outP + kV;
  float* outST    = outRL + 1;
  float* outSigma = outST + (size_t)kT * kV;

  zero_acc<<<2, 256, 0, stream>>>(gS1, kT + 1);
  precompute<<<(kT * kD + kD * kT) / 256, 256, 0, stream>>>(B, A, t2v, Mt, Qt);
  const int nTiles = (kV + 15) / 16;
  fused_main<<<nTiles, 128, 0, stream>>>(w2v, Mt, Qt, sigma,
                                         outAlpha, outP, outST, gS1, gSumSq);
  finalize<<<1, 256, 0, stream>>>(gS1, gSumSq, sigma, outRL, outSigma);
}
